// ContrastiveMROADMultiQueue_24103356465342
// MI455X (gfx1250) — matrix operand order repaired and checked by reading, not compiled, not run on hardware
//
#include <hip/hip_runtime.h>
#include <hip/hip_bf16.h>

// ---------------------------------------------------------------------------
// ContrastiveMROADMultiQueue for MI455X (gfx1250, wave32, WMMA).
// All large GEMMs run through v_wmma_f32_16x16x32_f16 (f16 A/B, f32 acc).
// GRU-recurrence GEMM stages its shared A tile via LDS (double-buffered) and
// prefetches the streamed whh weight pack (global_prefetch_b8).
// ---------------------------------------------------------------------------

typedef __attribute__((ext_vector_type(16))) _Float16 v16h;
typedef __attribute__((ext_vector_type(8)))  _Float16 v8h;
typedef __attribute__((ext_vector_type(8)))  float    v8f;

#define B_    32
#define T_    128
#define DRGB  2048
#define DIN   4096
#define E_    1024
#define H_    2048
#define H3    6144
#define C_    22
#define K_    1024
#define CD_   128
#define MOM   0.999f
#define MRAT  0.25f
#define EPS_  1e-5f

// ---- WMMA fragment helpers (16x16x32 f16) ----------------------------------
// A (16x32, row-major source): lane holds row (lane&15); elements map to
//   K = (i%8) + (i/8)*16 + (lane/16)*8  -> two contiguous 8-half loads.
__device__ __forceinline__ v16h load_a_frag(const _Float16* A, int lda,
                                            int row, int k0, int lane) {
  union { v16h v; v8h h[2]; } u;
  const _Float16* p = A + (size_t)row * lda + k0 + ((lane >> 4) << 3);
  u.h[0] = *(const v8h*)p;
  u.h[1] = *(const v8h*)(p + 16);
  return u.v;
}

// Same fragment gather out of a 32x32 LDS tile (row-major, ld=32).
__device__ __forceinline__ v16h lds_a_frag(const _Float16* At, int row, int lane) {
  union { v16h v; v8h h[2]; } u;
  const _Float16* p = At + row * 32 + ((lane >> 4) << 3);
  u.h[0] = *(const v8h*)p;
  u.h[1] = *(const v8h*)(p + 16);
  return u.v;
}

// B pre-packed per (ktile, ntile, lane): one contiguous 32B load per lane.
__device__ __forceinline__ v16h load_b_frag(const _Float16* Bp, int nTiles,
                                            int kt, int nt, int lane) {
  return *(const v16h*)(Bp + ((((size_t)kt * nTiles + nt) * 32 + lane) << 4));
}

// B (32x16) element mapping used by the packer: lane gives N=(lane&15),
// K = (i%8) + (i/8)*16 + (lane/16)*8  (matches A fragment order).
__device__ __forceinline__ int kmap_b(int i, int lane) {
  return (i & 7) + ((i >> 3) << 4) + ((lane >> 4) << 3);
}

// ---- weight packer: f32 (optionally momentum-blended, optionally transposed)
// -> f16 in per-lane WMMA-B layout --------------------------------------------
__global__ void pack_b_kernel(const float* __restrict__ Wk,
                              const float* __restrict__ Wq,  // null -> no blend
                              _Float16* __restrict__ out,
                              int Ktot, int Ntot, int ldw, int trans) {
  size_t tid = (size_t)blockIdx.x * blockDim.x + threadIdx.x;
  size_t total = (size_t)Ktot * Ntot;
  if (tid >= total) return;
  int i     = (int)(tid & 15);
  int lane  = (int)((tid >> 4) & 31);
  size_t tl = tid >> 9;                 // 512 elements per 32x16 tile
  int nTiles = Ntot >> 4;
  int nt = (int)(tl % nTiles);
  int kt = (int)(tl / nTiles);
  int k = (kt << 5) + kmap_b(i, lane);
  int n = (nt << 4) + (lane & 15);
  size_t src = trans ? ((size_t)n * ldw + k) : ((size_t)k * ldw + n);
  float v = Wk[src];
  if (Wq) v = MOM * v + (1.0f - MOM) * Wq[src];
  out[tid] = (_Float16)v;
}

__global__ void blend_kernel(const float* __restrict__ a,  // key
                             const float* __restrict__ b,  // query
                             float* __restrict__ dst, int n) {
  int tid = blockIdx.x * blockDim.x + threadIdx.x;
  if (tid < n) dst[tid] = MOM * a[tid] + (1.0f - MOM) * b[tid];
}

// ---- build concatenated (masked) input, f32 -> f16 --------------------------
__global__ void build_x_kernel(const float* __restrict__ rgb,
                               const float* __restrict__ flow,
                               const float* __restrict__ rmask,
                               _Float16* __restrict__ X, int masked) {
  size_t tid = (size_t)blockIdx.x * blockDim.x + threadIdx.x;
  if (tid >= (size_t)B_ * T_ * DIN) return;
  int col = (int)(tid & (DIN - 1));
  int row = (int)(tid >> 12);           // b*T + t
  int t   = row & (T_ - 1);
  float m = 1.0f;
  if (masked) {
    m = (rmask[row] > MRAT) ? 1.0f : 0.0f;
    if (t == T_ - 1) m = 1.0f;
  }
  float v = (col < DRGB) ? rgb[(size_t)row * DRGB + col]
                         : flow[(size_t)row * DRGB + (col - DRGB)];
  X[tid] = (_Float16)(v * m);
}

// ---- large GEMM: 128x128 block tile, 8 waves, 2x4 (MxN) 16x16 tiles/wave ----
// bias must be non-null (all call sites have one) -> branch-free epilogue.
__global__ __launch_bounds__(256)
void gemm_f16_kernel(const _Float16* __restrict__ A,
                     const _Float16* __restrict__ Bp,
                     const float* __restrict__ bias,
                     float* __restrict__ Cout,
                     int Ntot, int Ktot) {
  int lane = threadIdx.x & 31;
  int wave = threadIdx.x >> 5;          // 0..7
  int wm = wave >> 1;                   // 0..3
  int wn = wave & 1;                    // 0..1
  int m0 = blockIdx.y * 128 + wm * 32;
  int n0 = blockIdx.x * 128 + wn * 64;
  int nTiles = Ntot >> 4;
  int ntBase = n0 >> 4;

  v8f zero = {0.f,0.f,0.f,0.f,0.f,0.f,0.f,0.f};
  v8f c[2][4];
#pragma unroll
  for (int a = 0; a < 2; ++a)
#pragma unroll
    for (int b = 0; b < 4; ++b) c[a][b] = zero;

  int ksteps = Ktot >> 5;
  for (int kt = 0; kt < ksteps; ++kt) {
    int k0 = kt << 5;
    // prefetch B two k-tiles ahead (streams once; lowers to global_prefetch_b8)
    if (kt + 2 < ksteps)
      __builtin_prefetch(
          Bp + ((((size_t)(kt + 2) * nTiles + ntBase) * 32 + lane) << 4), 0, 1);
    v16h af[2];
    af[0] = load_a_frag(A, Ktot, m0 +      (lane & 15), k0, lane);
    af[1] = load_a_frag(A, Ktot, m0 + 16 + (lane & 15), k0, lane);
    v16h bf[4];
#pragma unroll
    for (int b = 0; b < 4; ++b)
      bf[b] = load_b_frag(Bp, nTiles, kt, ntBase + b, lane);
#pragma unroll
    for (int a = 0; a < 2; ++a)
#pragma unroll
      for (int b = 0; b < 4; ++b)
        c[a][b] = __builtin_amdgcn_wmma_f32_16x16x32_f16(
            false, af[a], false, bf[b], (short)0, c[a][b], false, false);
  }

  int col0 = lane & 15;
  int rowOff = (lane >> 4) << 3;
#pragma unroll
  for (int a = 0; a < 2; ++a)
#pragma unroll
    for (int b = 0; b < 4; ++b) {
      int n = n0 + b * 16 + col0;
      float bv = bias[n];
#pragma unroll
      for (int j = 0; j < 8; ++j) {
        int m = m0 + a * 16 + rowOff + j;
        Cout[(size_t)m * Ntot + n] = c[a][b][j] + bv;
      }
    }
}

// ---- small-M GEMM for GRU recurrence: M=32, block covers 32x64 --------------
// A (32 x Ktot) is shared by all 4 waves: stage each 32x32 k-tile through a
// double-buffered LDS tile so it is fetched from global exactly once/block.
__global__ __launch_bounds__(128)
void gemm_small_kernel(const _Float16* __restrict__ A,   // [32, Ktot] f16
                       const _Float16* __restrict__ Bp,
                       const float* __restrict__ bias,
                       float* __restrict__ Cout,         // [32, Ntot] f32
                       int Ntot, int Ktot) {
  __shared__ _Float16 As[2][32 * 32];
  int tid  = threadIdx.x;               // 0..127
  int lane = tid & 31;
  int wave = tid >> 5;                  // 0..3
  int n0 = blockIdx.x * 64 + wave * 16;
  int nt = n0 >> 4;
  int nTiles = Ntot >> 4;

  int srow = tid >> 2;                  // 0..31  (stager: one b128 per thread)
  int schk = (tid & 3) << 3;            // 0,8,16,24 halves

  v8f zero = {0.f,0.f,0.f,0.f,0.f,0.f,0.f,0.f};
  v8f c0 = zero, c1 = zero;

  int ksteps = Ktot >> 5;
  // stage k-tile 0
  *(v8h*)&As[0][srow * 32 + schk] =
      *(const v8h*)(A + (size_t)srow * Ktot + schk);
  __syncthreads();

  for (int kt = 0; kt < ksteps; ++kt) {
    int buf = kt & 1;
    if (kt + 1 < ksteps)                // stage next tile into other buffer
      *(v8h*)&As[buf ^ 1][srow * 32 + schk] =
          *(const v8h*)(A + (size_t)srow * Ktot + ((kt + 1) << 5) + schk);
    if (kt + 4 < ksteps)                // prefetch streamed B (24 MB whh pack)
      __builtin_prefetch(
          Bp + ((((size_t)(kt + 4) * nTiles + nt) * 32 + lane) << 4), 0, 1);

    v16h a0 = lds_a_frag(As[buf],      (lane & 15), lane);
    v16h a1 = lds_a_frag(As[buf], 16 + (lane & 15), lane);
    v16h bf = load_b_frag(Bp, nTiles, kt, nt, lane);
    c0 = __builtin_amdgcn_wmma_f32_16x16x32_f16(false, a0, false, bf, (short)0, c0, false, false);
    c1 = __builtin_amdgcn_wmma_f32_16x16x32_f16(false, a1, false, bf, (short)0, c1, false, false);
    __syncthreads();
  }

  int col0 = lane & 15;
  int rowOff = (lane >> 4) << 3;
  int n = n0 + col0;
  float bv = bias[n];
#pragma unroll
  for (int j = 0; j < 8; ++j) {
    Cout[(size_t)(rowOff + j) * Ntot + n]      = c0[j] + bv;
    Cout[(size_t)(rowOff + j + 16) * Ntot + n] = c1[j] + bv;
  }
}

// ---- LayerNorm + ReLU over rows of E=1024, f32 in -> f16 out ----------------
__global__ __launch_bounds__(256)
void ln_relu_kernel(const float* __restrict__ Y,
                    const float* __restrict__ g,
                    const float* __restrict__ be,
                    _Float16* __restrict__ out) {
  __shared__ float red[256];
  int row = blockIdx.x;
  const float* y = Y + (size_t)row * E_;
  float s = 0.f, s2 = 0.f;
  for (int i = threadIdx.x; i < E_; i += 256) {
    float v = y[i]; s += v; s2 += v * v;
  }
  red[threadIdx.x] = s; __syncthreads();
  for (int o = 128; o > 0; o >>= 1) {
    if (threadIdx.x < o) red[threadIdx.x] += red[threadIdx.x + o];
    __syncthreads();
  }
  float mu = red[0] * (1.0f / E_); __syncthreads();
  red[threadIdx.x] = s2; __syncthreads();
  for (int o = 128; o > 0; o >>= 1) {
    if (threadIdx.x < o) red[threadIdx.x] += red[threadIdx.x + o];
    __syncthreads();
  }
  float var = red[0] * (1.0f / E_) - mu * mu;
  float rstd = rsqrtf(var + EPS_);
  for (int i = threadIdx.x; i < E_; i += 256) {
    float v = (y[i] - mu) * rstd * g[i] + be[i];
    out[(size_t)row * E_ + i] = (_Float16)(v > 0.f ? v : 0.f);
  }
}

// ---- GRU gates (torch order r|z|n); updates h in place + f16 copy -----------
__global__ void gru_gates_kernel(const float* __restrict__ xi,
                                 const float* __restrict__ hh,
                                 float* __restrict__ h,
                                 _Float16* __restrict__ h16, int t) {
  int tid = blockIdx.x * blockDim.x + threadIdx.x;
  if (tid >= B_ * H_) return;
  int b = tid >> 11;
  int c = tid & (H_ - 1);
  const float* xr = xi + (size_t)(b * T_ + t) * H3;
  const float* hr = hh + (size_t)b * H3;
  float r = 1.f / (1.f + __expf(-(xr[c] + hr[c])));
  float z = 1.f / (1.f + __expf(-(xr[c + H_] + hr[c + H_])));
  float n = tanhf(xr[c + 2 * H_] + r * hr[c + 2 * H_]);
  float hp = h[tid];
  float hn = (1.f - z) * n + z * hp;
  h[tid] = hn;
  h16[tid] = (_Float16)hn;
}

__global__ void zero_h_kernel(float* h, _Float16* h16) {
  int tid = blockIdx.x * blockDim.x + threadIdx.x;
  if (tid < B_ * H_) { h[tid] = 0.f; h16[tid] = (_Float16)0.f; }
}

__global__ void relu_copy_kernel(const float* __restrict__ h,
                                 float* __restrict__ feat) {
  int tid = blockIdx.x * blockDim.x + threadIdx.x;
  if (tid < B_ * H_) feat[tid] = fmaxf(h[tid], 0.f);
}

// ---- projection + L2 normalize: out[b] = l2norm(feat[b] @ wq + bq) ----------
__global__ __launch_bounds__(128)
void proj_norm_kernel(const float* __restrict__ feat,
                      const float* __restrict__ wq,
                      const float* __restrict__ bq,
                      float* __restrict__ out) {
  __shared__ float red[128];
  int b = blockIdx.x;
  int d = threadIdx.x;
  const float* f = feat + (size_t)b * H_;
  float acc = bq[d];
  for (int k = 0; k < H_; ++k) acc = fmaf(f[k], wq[(size_t)k * CD_ + d], acc);
  red[d] = acc * acc; __syncthreads();
  for (int o = 64; o > 0; o >>= 1) {
    if (d < o) red[d] += red[d + o];
    __syncthreads();
  }
  out[(size_t)b * CD_ + d] = acc * rsqrtf(red[0]);
}

// ---- per-class circular FIFO enqueue (queues already copied to out) ---------
__global__ __launch_bounds__(128)
void queue_scatter_kernel(const float* __restrict__ targets,
                          const int* __restrict__ ptrs,
                          const float* __restrict__ k_cls,
                          float* __restrict__ qout,
                          int* __restrict__ ptrs_out) {
  int c = blockIdx.x;
  int d = threadIdx.x;
  int ptr = ptrs[c];
  int cnt = 0;
  for (int b = 0; b < B_; ++b) {
    if (targets[b * C_ + c] > 0.5f) {
      int slot = (ptr + cnt) % K_;
      qout[((size_t)c * CD_ + d) * K_ + slot] = k_cls[(size_t)b * CD_ + d];
      cnt++;
    }
  }
  if (d == 0) ptrs_out[c] = (ptr + cnt) % K_;
}

// ---------------------------------------------------------------------------
extern "C" void kernel_launch(void* const* d_in, const int* in_sizes, int n_in,
                              void* d_out, int out_size, void* d_ws, size_t ws_size,
                              hipStream_t stream) {
  const float* rgb   = (const float*)d_in[0];
  const float* flow  = (const float*)d_in[1];
  const float* rmask = (const float*)d_in[2];
  const float* targs = (const float*)d_in[3];
  const float* w1_q  = (const float*)d_in[4];
  const float* b1_q  = (const float*)d_in[5];
  const float* g_q   = (const float*)d_in[6];
  const float* be_q  = (const float*)d_in[7];
  const float* wih_q = (const float*)d_in[8];
  const float* whh_q = (const float*)d_in[9];
  const float* bih_q = (const float*)d_in[10];
  const float* bhh_q = (const float*)d_in[11];
  const float* w1_k  = (const float*)d_in[12];
  const float* b1_k  = (const float*)d_in[13];
  const float* g_k   = (const float*)d_in[14];
  const float* be_k  = (const float*)d_in[15];
  const float* wih_k = (const float*)d_in[16];
  const float* whh_k = (const float*)d_in[17];
  const float* bih_k = (const float*)d_in[18];
  const float* bhh_k = (const float*)d_in[19];
  const float* wqw   = (const float*)d_in[20];
  const float* bqb   = (const float*)d_in[21];
  const int*   ptrs  = (const int*)d_in[22];
  const float* ques  = (const float*)d_in[23];

  // --- bump allocator over workspace ---
  char* ws = (char*)d_ws;
  size_t off = 0;
  auto alloc = [&](size_t bytes) -> void* {
    void* p = ws + off;
    off += (bytes + 255) & ~(size_t)255;
    return p;
  };
  _Float16* X16   = (_Float16*)alloc((size_t)B_ * T_ * DIN * 2);   // 32 MB
  _Float16* W1p   = (_Float16*)alloc((size_t)DIN * E_ * 2);        // 8 MB
  _Float16* WihTp = (_Float16*)alloc((size_t)E_ * H3 * 2);         // 12 MB
  _Float16* WhhTp = (_Float16*)alloc((size_t)H_ * H3 * 2);         // 24 MB
  float*    y1    = (float*)   alloc((size_t)B_ * T_ * E_ * 4);    // 16 MB
  _Float16* xln   = (_Float16*)alloc((size_t)B_ * T_ * E_ * 2);    // 8 MB
  float*    xi    = (float*)   alloc((size_t)B_ * T_ * H3 * 4);    // 100 MB
  float*    hh    = (float*)   alloc((size_t)B_ * H3 * 4);
  float*    h     = (float*)   alloc((size_t)B_ * H_ * 4);
  _Float16* h16   = (_Float16*)alloc((size_t)B_ * H_ * 2);
  float*    featq = (float*)   alloc((size_t)B_ * H_ * 4);
  float*    featk = (float*)   alloc((size_t)B_ * H_ * 4);
  float*    kbias = (float*)   alloc((size_t)(3 * E_ + 2 * H3) * 4);
  float* kb1  = kbias;
  float* kg   = kbias + E_;
  float* kbe  = kbias + 2 * E_;
  float* kbih = kbias + 3 * E_;
  float* kbhh = kbias + 3 * E_ + H3;

  const int BT = B_ * T_;
  float* out_f = (float*)d_out;

  for (int enc = 0; enc < 2; ++enc) {
    const float *w1, *wih, *whh, *w1a, *wiha, *whha;
    const float *b1, *g, *be, *bih, *bhh;
    if (enc == 0) {
      w1 = w1_q; wih = wih_q; whh = whh_q; w1a = nullptr; wiha = nullptr; whha = nullptr;
      b1 = b1_q; g = g_q; be = be_q; bih = bih_q; bhh = bhh_q;
    } else {
      w1 = w1_k; wih = wih_k; whh = whh_k; w1a = w1_q; wiha = wih_q; whha = whh_q;
      // blended key biases
      blend_kernel<<<(E_ + 255) / 256, 256, 0, stream>>>(b1_k, b1_q, kb1, E_);
      blend_kernel<<<(E_ + 255) / 256, 256, 0, stream>>>(g_k,  g_q,  kg,  E_);
      blend_kernel<<<(E_ + 255) / 256, 256, 0, stream>>>(be_k, be_q, kbe, E_);
      blend_kernel<<<(H3 + 255) / 256, 256, 0, stream>>>(bih_k, bih_q, kbih, H3);
      blend_kernel<<<(H3 + 255) / 256, 256, 0, stream>>>(bhh_k, bhh_q, kbhh, H3);
      b1 = kb1; g = kg; be = kbe; bih = kbih; bhh = kbhh;
    }

    // pack weights (momentum-blended for key) into WMMA-B layout (f16)
    {
      size_t n1 = (size_t)DIN * E_;
      pack_b_kernel<<<(unsigned)((n1 + 255) / 256), 256, 0, stream>>>(
          w1, w1a, W1p, DIN, E_, E_, 0);                     // B[k][n] = w1[k][n]
      size_t n2 = (size_t)E_ * H3;
      pack_b_kernel<<<(unsigned)((n2 + 255) / 256), 256, 0, stream>>>(
          wih, wiha, WihTp, E_, H3, E_, 1);                  // B[k][n] = wih[n][k]
      size_t n3 = (size_t)H_ * H3;
      pack_b_kernel<<<(unsigned)((n3 + 255) / 256), 256, 0, stream>>>(
          whh, whha, WhhTp, H_, H3, H_, 1);                  // B[k][n] = whh[n][k]
    }

    // input assembly (masked for query, plain for key)
    {
      size_t n = (size_t)BT * DIN;
      build_x_kernel<<<(unsigned)((n + 255) / 256), 256, 0, stream>>>(
          rgb, flow, rmask, X16, enc == 0 ? 1 : 0);
    }

    // layer1: X @ W1 + b1  -> y1, then LN+ReLU -> xln (f16)
    gemm_f16_kernel<<<dim3(E_ / 128, BT / 128), 256, 0, stream>>>(
        X16, W1p, b1, y1, E_, DIN);
    ln_relu_kernel<<<BT, 256, 0, stream>>>(y1, g, be, xln);

    // GRU input projection: xln @ wih.T + bih -> xi [B*T, 3H]
    gemm_f16_kernel<<<dim3(H3 / 128, BT / 128), 256, 0, stream>>>(
        xln, WihTp, bih, xi, H3, E_);

    // GRU recurrence
    zero_h_kernel<<<(B_ * H_ + 255) / 256, 256, 0, stream>>>(h, h16);
    for (int t = 0; t < T_; ++t) {
      gemm_small_kernel<<<H3 / 64, 128, 0, stream>>>(h16, WhhTp, bhh, hh, H3, H_);
      gru_gates_kernel<<<(B_ * H_ + 255) / 256, 256, 0, stream>>>(xi, hh, h, h16, t);
    }
    relu_copy_kernel<<<(B_ * H_ + 255) / 256, 256, 0, stream>>>(
        h, enc == 0 ? featq : featk);

    // projection + l2norm -> q_cls / k_cls region of d_out
    proj_norm_kernel<<<B_, 128, 0, stream>>>(
        enc == 0 ? featq : featk, wqw, bqb,
        out_f + (size_t)enc * B_ * CD_);
  }

  // queues: copy then per-class FIFO scatter of k_cls; ptr update
  float* qout = out_f + 2 * (size_t)B_ * CD_;
  hipMemcpyAsync(qout, ques, (size_t)C_ * CD_ * K_ * sizeof(float),
                 hipMemcpyDeviceToDevice, stream);
  int* ptrs_out = (int*)(qout + (size_t)C_ * CD_ * K_);
  queue_scatter_kernel<<<C_, 128, 0, stream>>>(
      targs, ptrs, out_f + (size_t)B_ * CD_, qout, ptrs_out);
}
